// RvNN_84911503442506
// MI455X (gfx1250) — matrix operand order repaired and checked by reading, not compiled, not run on hardware
//
#include <hip/hip_runtime.h>
#include <hip/hip_bf16.h>

// ---------------------------------------------------------------------------
// Tree-GRU (RvNN) for MI455X / gfx1250.  B=8, N=512, D=256.
//   Phase A (parallel): pre[b,n,0:768] = h_e @ [W_ioux;W_coux]^T + folded
//     biases via v_wmma_f32_16x16x32_bf16 (all operands pre-converted bf16).
//   Phase B (serial, 512 steps): one workgroup per batch; hidden state lives
//     in 256KB of LDS as bf16 (CDNA5 WGP has 320KB). Initial state loaded by
//     the Tensor Data Mover (tensor_load_to_lds, TENSORcnt); per-step
//     adjacency row staged with global_load_async_to_lds_b32 (ASYNCcnt).
//     Per-step GEMVs run as M-padded 16x16x32 bf16 WMMAs.
// Workspace layout (bytes), ~19.7 MB total:
//   pre      f32  [4096][768]   @ 0         (12,582,912)
//   normT    bf16 [8][512][512] @ 12582912  ( 4,194,304)
//   Wt_iouh  bf16 [256][512]    @ 16777216  (   262,144)
//   Wt_couh  bf16 [256][256]    @ 17039360  (   131,072)
//   Wt_x     bf16 [256][768]    @ 17170432  (   393,216)
//   hbf      bf16 [8][512][256] @ 17563648  ( 2,097,152)
//   dinv     f32  [8][512]      @ 19660800  (    16,384)
// ---------------------------------------------------------------------------

typedef __attribute__((ext_vector_type(16))) __bf16    v16bf;
typedef __attribute__((ext_vector_type(8)))  __bf16    v8bf;
typedef __attribute__((ext_vector_type(8)))  float     v8f;
typedef __attribute__((ext_vector_type(4)))  unsigned  u32x4;
typedef __attribute__((ext_vector_type(8)))  unsigned  u32x8;

__device__ __forceinline__ v8f wmma_bf16(v16bf a, v16bf b, v8f c) {
  return __builtin_amdgcn_wmma_f32_16x16x32_bf16(false, a, false, b, (short)0, c,
                                                 false, false);
}

// Concatenate two contiguous 8-element bf16 chunks into one A/B fragment.
__device__ __forceinline__ v16bf frag_cat(const __bf16* p) {
  v8bf lo = *(const v8bf*)(p);
  v8bf hi = *(const v8bf*)(p + 16);
  return __builtin_shufflevector(lo, hi, 0, 1, 2, 3, 4, 5, 6, 7,
                                 8, 9, 10, 11, 12, 13, 14, 15);
}

// ---------------- degree inverse: dinv[b,i] = 1/sum_j adj[b,i,j] -----------
__global__ __launch_bounds__(256) void k_dinv(const float* __restrict__ adj,
                                              float* __restrict__ dinv) {
  const int lane = threadIdx.x & 31;
  const int row  = blockIdx.x * 8 + (threadIdx.x >> 5);  // 0..4095 = b*512+i
  const float* p = adj + (size_t)row * 512;
  float s = 0.f;
#pragma unroll
  for (int t = 0; t < 16; ++t) s += p[lane + 32 * t];
#pragma unroll
  for (int off = 16; off > 0; off >>= 1) s += __shfl_xor(s, off, 32);
  if (lane == 0) dinv[row] = (s != 0.f) ? (1.f / s) : 0.f;
}

// ------------- normT[b,i,j] = bf16(adj[b,j,i] * dinv[b,i]) -----------------
__global__ __launch_bounds__(256) void k_normT(const float* __restrict__ adj,
                                               const float* __restrict__ dinv,
                                               __bf16* __restrict__ normT) {
  const size_t t = (size_t)blockIdx.x * 256 + threadIdx.x;  // < 2M
  const int j  = (int)(t & 511);
  const int bi = (int)(t >> 9);
  const int i  = bi & 511;
  const int b  = bi >> 9;
  normT[t] = (__bf16)(adj[((size_t)b * 512 + j) * 512 + i] * dinv[bi]);
}

// ------------- h_e -> bf16 -------------------------------------------------
__global__ __launch_bounds__(256) void k_hcvt(const float* __restrict__ he,
                                              __bf16* __restrict__ hbf) {
  const size_t t = (size_t)blockIdx.x * 256 + threadIdx.x;  // < 2M
  hbf[t] = (__bf16)he[t];
}

// ------------- transpose all weights to bf16, k-major ----------------------
__global__ __launch_bounds__(256) void k_wcvt(const float* __restrict__ Wiouh,
                                              const float* __restrict__ Wcouh,
                                              const float* __restrict__ Wioux,
                                              const float* __restrict__ Wcoux,
                                              __bf16* __restrict__ Wt_iouh,
                                              __bf16* __restrict__ Wt_couh,
                                              __bf16* __restrict__ Wt_x) {
  const int t = blockIdx.x * 256 + threadIdx.x;   // < 393216
  if (t < 131072) {                                // Wt_iouh[k][n] = W_iouh[n][k]
    const int k = t / 512, n = t % 512;
    Wt_iouh[t] = (__bf16)Wiouh[(size_t)n * 256 + k];
  } else if (t < 196608) {                         // Wt_couh[k][n] = W_couh[n][k]
    const int t2 = t - 131072;
    const int k = t2 / 256, n = t2 % 256;
    Wt_couh[t2] = (__bf16)Wcouh[(size_t)n * 256 + k];
  } else {                                         // Wt_x[k][c] = Wcat[c][k]
    const int t3 = t - 196608;
    const int k = t3 / 768, c = t3 % 768;
    Wt_x[t3] = (__bf16)((c < 512) ? Wioux[(size_t)c * 256 + k]
                                  : Wcoux[(size_t)(c - 512) * 256 + k]);
  }
}

// ------------- pre = h_e @ [W_ioux;W_coux]^T + (b_*x + b_*h) ---------------
// One 16x16 output tile per wave; M=4096 (B*N), Nout=768, K=256.
__global__ __launch_bounds__(256) void k_gemm_pre(
    const __bf16* __restrict__ hbf, const __bf16* __restrict__ Wt_x,
    const float* __restrict__ bioux, const float* __restrict__ biouh,
    const float* __restrict__ bcoux, const float* __restrict__ bcouh,
    float* __restrict__ pre) {
  const int lane  = threadIdx.x & 31;
  const int gw    = blockIdx.x * 8 + (threadIdx.x >> 5);
  const int mtile = gw / 48, ntile = gw % 48;
  const int hi    = lane >> 4;
  const int mrow  = mtile * 16 + (lane & 15);

  v8f acc = {};
#pragma unroll
  for (int kt = 0; kt < 8; ++kt) {
    // A frag: lane m holds K = kt*32 + hi*8 + {0..7, 16..23}
    v16bf a = frag_cat(hbf + (size_t)mrow * 256 + kt * 32 + hi * 8);
    // B frag: lane n holds K = kt*32 + hi*8 + {0..7, 16..23}
    v16bf bb;
#pragma unroll
    for (int s = 0; s < 16; ++s)
      bb[s] = Wt_x[(size_t)(kt * 32 + hi * 8 + (s & 7) + (s >> 3) * 16) * 768 + ntile * 16 + (lane & 15)];
    acc = wmma_bf16(a, bb, acc);
  }
  const int c = ntile * 16 + (lane & 15);
  const float bias = (c < 512) ? (bioux[c] + biouh[c])
                               : (bcoux[c - 512] + bcouh[c - 512]);
#pragma unroll
  for (int v = 0; v < 8; ++v) {
    const int row = mtile * 16 + v + hi * 8;
    pre[(size_t)row * 768 + c] = acc[v] + bias;
  }
}

// ------------- serial scan: one workgroup (16 waves) per batch -------------
// Dynamic-LDS layout (no static __shared__, so dynamic LDS starts at byte 0):
//   [0      , 262144) h_lds   bf16[512][256]
//   [262144 , 263168) row_bf  bf16[512]
//   [263168 , 264192) xpar_f  f32[256]
//   [264192 , 264704) xpar_bf bf16[256]
//   [264704 , 265216) xr_bf   bf16[256]
//   [265216 , 266240) z_f     f32[256]
__global__ __launch_bounds__(512) void k_seq(
    const __bf16* __restrict__ hbf, const __bf16* __restrict__ normT,
    const __bf16* __restrict__ Wt_iouh, const __bf16* __restrict__ Wt_couh,
    const float* __restrict__ pre, float* __restrict__ out) {
  extern __shared__ char smem[];
  __bf16* h_lds   = (__bf16*)smem;
  __bf16* row_bf  = (__bf16*)(smem + 262144);
  float*  xpar_f  = (float*)(smem + 263168);
  __bf16* xpar_bf = (__bf16*)(smem + 264192);
  __bf16* xr_bf   = (__bf16*)(smem + 264704);
  float*  z_f     = (float*)(smem + 265216);

  const int b    = blockIdx.x;
  const int tid  = threadIdx.x;
  const int lane = tid & 31, w = tid >> 5;
  const int hi   = lane >> 4;

  // ---- Initial hidden state via Tensor Data Mover: one 2D bf16 tile,
  //      512 rows x 256 cols, global (hbf + b*256KB) -> LDS offset 0.
  if (w == 0) {
    const unsigned long long ga =
        (unsigned long long)(hbf + (size_t)b * 512 * 256);   // byte address
    u32x4 g0;
    g0[0] = 1u;                                   // count=1, user descriptor
    g0[1] = 0u;                                   // lds_addr = 0
    g0[2] = (unsigned)ga;                         // global_addr[31:0]
    g0[3] = ((unsigned)(ga >> 32) & 0x01FFFFFFu)  // global_addr[56:32]
            | (2u << 30);                         // type = 2 ("image")
    u32x8 g1;
    g1[0] = 0x00010000u;       // data_size = 1 (2 bytes/elem)
    g1[1] = 256u << 16;        // tensor_dim0 = 256 (bits 79:48, low half)
    g1[2] = 512u << 16;        // tensor_dim1 = 512 (bits 111:80, low half)
    g1[3] = 256u << 16;        // tile_dim0   = 256 (bits 127:112)
    g1[4] = 512u;              // tile_dim1   = 512 (bits 143:128)
    g1[5] = 256u;              // tensor_dim0_stride = 256 (bits 207:160)
    g1[6] = 0u;
    g1[7] = 0u;
    asm volatile("tensor_load_to_lds %0, %1" :: "s"(g0), "s"(g1) : "memory");
    __builtin_amdgcn_s_wait_tensorcnt(0);
  }
  __syncthreads();

  const unsigned long long nrow_base0 =
      (unsigned long long)(normT + (size_t)b * 512 * 512);

  for (int i = 0; i < 512; ++i) {
    // ---- Stage normalized-transposed adjacency row i (1KB) asynchronously
    //      into LDS: 256 lanes x b32, GVS addressing (SGPR base + VGPR off).
    if (tid < 256) {
      const unsigned lds_addr = 262144u + (unsigned)tid * 4u;  // row_bf
      const unsigned goff     = (unsigned)tid * 4u;
      const unsigned long long base = nrow_base0 + (unsigned long long)i * 1024u;
      asm volatile("global_load_async_to_lds_b32 %0, %1, %2"
                   :: "v"(lds_addr), "v"(goff), "s"(base) : "memory");
    }
    if (i + 1 < 512 && tid < 12)   // warm next step's pre row (global_prefetch_b8)
      __builtin_prefetch(pre + ((size_t)b * 512 + (i + 1)) * 768 + tid * 64, 0, 1);
    asm volatile("s_wait_asynccnt 0x0" ::: "memory");
    __syncthreads();

    // ---- Phase 1: x_par = rowvec @ h (K=512, LDS h); wave w -> cols [16w,16w+16)
    {
      const int nbase = w * 16;
      v8f acc = {};
#pragma unroll
      for (int kt = 0; kt < 16; ++kt) {
        v16bf a = {};                              // only M row 0 is live
        if ((lane & 15) == 0) a = frag_cat(row_bf + kt * 32 + hi * 8);
        v16bf hb;                                  // B frag: lane -> N, elems -> K
#pragma unroll
        for (int s = 0; s < 16; ++s)
          hb[s] = h_lds[(kt * 32 + hi * 8 + (s & 7) + (s >> 3) * 16) * 256 + nbase + (lane & 15)];
        acc = wmma_bf16(a, hb, acc);
      }
      if (lane < 16) {                             // M=0 row: VGPR0, lanes 0-15
        xpar_f[nbase + lane]  = acc[0];
        xpar_bf[nbase + lane] = (__bf16)acc[0];
      }
    }
    __syncthreads();

    // ---- Phase 2: iou = pre_iou + x_par @ W_iouh^T ; gates r/z
#pragma unroll
    for (int part = 0; part < 2; ++part) {
      const int nt = part * 16 + w;                // part 0 -> r, part 1 -> z
      v8f acc = {};
#pragma unroll
      for (int kt = 0; kt < 8; ++kt) {
        v16bf a = {};
        if ((lane & 15) == 0) a = frag_cat(xpar_bf + kt * 32 + hi * 8);
        v16bf wb;
#pragma unroll
        for (int s = 0; s < 16; ++s)
          wb[s] = Wt_iouh[(size_t)(kt * 32 + hi * 8 + (s & 7) + (s >> 3) * 16) * 512 + nt * 16 + (lane & 15)];
        acc = wmma_bf16(a, wb, acc);
      }
      if (lane < 16) {
        const int n = nt * 16 + lane;
        const float v = acc[0] + pre[((size_t)b * 512 + i) * 768 + n];
        const float s = 1.f / (1.f + __expf(-v));
        if (n < 256) xr_bf[n] = (__bf16)(xpar_f[n] * s);  // x_par * r
        else         z_f[n - 256] = s;                    // z
      }
    }
    __syncthreads();

    // ---- Phase 3: hc = tanh(pre_c + (x_par*r) @ W_couh^T); h = z*x_par+(1-z)*hc
    {
      const int nbase = w * 16;
      v8f acc = {};
#pragma unroll
      for (int kt = 0; kt < 8; ++kt) {
        v16bf a = {};
        if ((lane & 15) == 0) a = frag_cat(xr_bf + kt * 32 + hi * 8);
        v16bf wb;
#pragma unroll
        for (int s = 0; s < 16; ++s)
          wb[s] = Wt_couh[(size_t)(kt * 32 + hi * 8 + (s & 7) + (s >> 3) * 16) * 256 + nbase + (lane & 15)];
        acc = wmma_bf16(a, wb, acc);
      }
      if (lane < 16) {
        const int n = nbase + lane;
        const float cv = acc[0] + pre[((size_t)b * 512 + i) * 768 + 512 + n];
        const float hc = tanhf(cv);
        const float z  = z_f[n];
        const float xp = xpar_f[n];
        const float hn = z * xp + (1.f - z) * hc;
        out[((size_t)b * 512 + i) * 256 + n] = hn;   // fp32 result
        h_lds[i * 256 + n] = (__bf16)hn;             // live state
      }
    }
    __syncthreads();
  }
}

// ---------------------------------------------------------------------------
extern "C" void kernel_launch(void* const* d_in, const int* in_sizes, int n_in,
                              void* d_out, int out_size, void* d_ws, size_t ws_size,
                              hipStream_t stream) {
  (void)in_sizes; (void)n_in; (void)out_size; (void)ws_size;
  const float* he    = (const float*)d_in[0];
  const float* adj   = (const float*)d_in[1];
  const float* Wioux = (const float*)d_in[2];
  const float* bioux = (const float*)d_in[3];
  const float* Wiouh = (const float*)d_in[4];
  const float* biouh = (const float*)d_in[5];
  const float* Wcoux = (const float*)d_in[6];
  const float* bcoux = (const float*)d_in[7];
  const float* Wcouh = (const float*)d_in[8];
  const float* bcouh = (const float*)d_in[9];
  float* out = (float*)d_out;

  char* ws = (char*)d_ws;
  float*  pre     = (float*)(ws);
  __bf16* normT   = (__bf16*)(ws + 12582912);
  __bf16* Wt_iouh = (__bf16*)(ws + 16777216);
  __bf16* Wt_couh = (__bf16*)(ws + 17039360);
  __bf16* Wt_x    = (__bf16*)(ws + 17170432);
  __bf16* hbf     = (__bf16*)(ws + 17563648);
  float*  dinv    = (float*)(ws + 19660800);

  k_dinv<<<512, 256, 0, stream>>>(adj, dinv);
  k_hcvt<<<8192, 256, 0, stream>>>(he, hbf);
  k_wcvt<<<1536, 256, 0, stream>>>(Wiouh, Wcouh, Wioux, Wcoux,
                                   Wt_iouh, Wt_couh, Wt_x);
  k_normT<<<8192, 256, 0, stream>>>(adj, dinv, normT);
  k_gemm_pre<<<1536, 256, 0, stream>>>(hbf, Wt_x, bioux, biouh, bcoux, bcouh,
                                       pre);
  // 266,240 B dynamic LDS: 256KB bf16 hidden state + staging buffers.
  k_seq<<<8, 512, 266240, stream>>>(hbf, normT, Wt_iouh, Wt_couh, pre, out);
}
